// multi_level_mlp_expert_55061480735368
// MI455X (gfx1250) — compile-verified, matrix operand order checked by reading, not run on hardware
//
#include <hip/hip_runtime.h>

// ---------------------------------------------------------------------------
// CDNA5 (gfx1250) types / WMMA helpers
// ---------------------------------------------------------------------------
typedef _Float16 h16;
typedef __attribute__((ext_vector_type(8)))  _Float16 v8h;
typedef __attribute__((ext_vector_type(16))) _Float16 v16h;
typedef __attribute__((ext_vector_type(8)))  float    v8f;

union Frag16 { v16h v; struct { v8h lo; v8h hi; } p; };

// Load a 16-half WMMA fragment.  Memory is [row][K] row-major with K
// contiguous; lane l supplies row (l&15).  Per ISA A/B layout, lane halves
// take K offsets {0,8} for the low 4 VGPRs and {16,24} for the high 4, so the
// caller passes p already offset by (lane>>4)*8 and we grab [0..7] + [16..23].
__device__ __forceinline__ v16h ldfrag(const h16* p) {
  Frag16 f;
  f.p.lo = *(const v8h*)(p);
  f.p.hi = *(const v8h*)(p + 16);
  return f.v;
}

__device__ __forceinline__ v8f wmma_f32_16x16x32(v16h a, v16h b, v8f c) {
  // (neg_a, A, neg_b, B, c_mod, C, reuse_a, reuse_b)
  return __builtin_amdgcn_wmma_f32_16x16x32_f16(false, a, false, b, (short)0, c,
                                                false, false);
}

// Async global->LDS 16B copy (per-lane), tracked by ASYNCcnt.
__device__ __forceinline__ void async_copy_b128(unsigned lds_off, const h16* gp) {
  asm volatile("global_load_async_to_lds_b128 %0, %1, off"
               :: "v"(lds_off), "v"(gp)
               : "memory");
}

__device__ __forceinline__ void wait_asynccnt0() {
  asm volatile("s_wait_asynccnt 0" ::: "memory");
}

__device__ __forceinline__ float waveAllSum(float v) {
#pragma unroll
  for (int off = 16; off > 0; off >>= 1) v += __shfl_xor(v, off, 32);
  return v;
}

// ---------------------------------------------------------------------------
// Problem constants
// ---------------------------------------------------------------------------
#define BATCH 4096
#define D1    10816   // 64*13*13
#define D2    128
#define NEXP  8
#define NOUT  10

// workspace byte offsets (all multiples of 128)
#define WS_ACC   ((size_t)0)           // 32 floats: imp1[8] load1[8] imp2[8] load2[8]
#define WS_BMAT  ((size_t)512)         // conv2 weights f16 [64][288]
#define WS_W1T   ((size_t)37376)       // W1 transposed f16 [1024][10816]
#define WS_H1    ((size_t)22188544)    // conv1 out f16 NHWC [4096][28][28][32]
#define WS_H2    ((size_t)227709440)   // pooled/flattened f16 [4096][10816]
#define WS_EOUT  ((size_t)316314112)   // expert outs f32 [4096][1024]
#define WS_GTS1  ((size_t)333091328)   // gates layer1 f32 [4096][8]
#define WS_Y1    ((size_t)333222400)   // moe1 output f32 [4096][128]

// ---------------------------------------------------------------------------
// K0: zero loss accumulators
// ---------------------------------------------------------------------------
__global__ void k_zero_acc(float* __restrict__ accs) {
  if (threadIdx.x < 32) accs[threadIdx.x] = 0.f;
}

// ---------------------------------------------------------------------------
// K1: conv1 (1->32, 3x3, valid) + relu, fp32 math, f16 NHWC output
// ---------------------------------------------------------------------------
__global__ void k_conv1(const float* __restrict__ x, const float* __restrict__ w,
                        const float* __restrict__ bias, h16* __restrict__ h1) {
  int tid = blockIdx.x * blockDim.x + threadIdx.x;  // [0, 4096*784*32)
  int c   = tid & 31;
  int pix = tid >> 5;
  int px  = pix % 28;
  int py  = (pix / 28) % 28;
  int b   = pix / 784;
  if (b >= BATCH) return;
  const float* in = x + (size_t)b * 900;  // 30*30
  float acc = bias[c];
#pragma unroll
  for (int ty = 0; ty < 3; ++ty)
#pragma unroll
    for (int tx = 0; tx < 3; ++tx)
      acc += in[(py + ty) * 30 + (px + tx)] * w[c * 9 + ty * 3 + tx];
  acc = fmaxf(acc, 0.f);
  h1[(size_t)pix * 32 + c] = (h16)acc;
}

// ---------------------------------------------------------------------------
// K2a: pack conv2 weights OIHW f32 -> [N=64][K=288] f16,  K = (ky*3+kx)*32 + c
// ---------------------------------------------------------------------------
__global__ void k_prep_bmat(const float* __restrict__ w2, h16* __restrict__ bm) {
  int idx = blockIdx.x * blockDim.x + threadIdx.x;
  if (idx >= 64 * 288) return;
  int o  = idx / 288;
  int kk = idx % 288;
  int kc = kk >> 5;        // tap index ky*3+kx
  int c  = kk & 31;        // input channel
  bm[idx] = (h16)w2[((o * 32 + c) * 3 + kc / 3) * 3 + (kc % 3)];
}

// ---------------------------------------------------------------------------
// K2b: W1 [8][10816][128] f32 -> W1t [1024][10816] f16   (n = e*128+o)
// ---------------------------------------------------------------------------
__global__ void k_prep_w1t(const float* __restrict__ W1, h16* __restrict__ w1t) {
  size_t idx = (size_t)blockIdx.x * blockDim.x + threadIdx.x;
  if (idx >= (size_t)1024 * D1) return;
  int i = (int)(idx % D1);
  int n = (int)(idx / D1);
  int e = n >> 7, o = n & 127;
  w1t[idx] = (h16)W1[((size_t)e * D1 + i) * 128 + o];
}

// ---------------------------------------------------------------------------
// K3: conv2 (32->64) as implicit GEMM WMMA + bias + relu + 2x2 maxpool fused.
// grid (43 pool-tiles, 4096 images), 128 threads = 4 waves; wave w owns output
// channels [16w,16w+16).  A-tile rows = 4 pooled cells x 4 quadrants so the
// maxpool reduces inside the accumulator (rows 0-3 / 4-7 per lane half).
// Output written in NCHW-flatten order: h2[b][c*169 + pooled_pos] f16.
// ---------------------------------------------------------------------------
__global__ void k_conv2pool(const h16* __restrict__ h1, const h16* __restrict__ bm,
                            const float* __restrict__ cbias, h16* __restrict__ h2) {
  int b     = blockIdx.y;
  int tileP = blockIdx.x;                 // 4 pooled cells per tile, 169 total
  int wave  = threadIdx.x >> 5;
  int lane  = threadIdx.x & 31;
  int mrow  = lane & 15;                  // A row this lane feeds
  int half  = lane >> 4;

  int pl = mrow >> 2;                     // pooled cell within tile (0..3)
  int q  = mrow & 3;                      // quadrant within 2x2
  int p  = tileP * 4 + pl; if (p > 168) p = 168;   // clamp (last tile pads)
  int py = p / 13, px = p % 13;
  int cy = 2 * py + (q >> 1);             // conv output position (0..25)
  int cx = 2 * px + (q & 1);

  const h16* abase = h1 + ((size_t)(b * 28 + cy) * 28 + cx) * 32 + half * 8;
  int n = wave * 16 + mrow;               // output channel for B row / C col
  const h16* bbase = bm + n * 288 + half * 8;

  v8f acc = {};
#pragma unroll
  for (int kc = 0; kc < 9; ++kc) {        // K = 9 taps * 32 channels
    int dy = kc / 3, dx = kc % 3;
    v16h a  = ldfrag(abase + (dy * 28 + dx) * 32);
    v16h bb = ldfrag(bbase + kc * 32);
    acc = wmma_f32_16x16x32(a, bb, acc);
  }

  // C layout: VGPR r = conv row r (lanes 0-15) / r+8 (lanes 16-31), col = lane&15
  float cb = cbias[n];
  float pa = fmaxf(fmaxf(acc[0], acc[1]), fmaxf(acc[2], acc[3]));
  float pb = fmaxf(fmaxf(acc[4], acc[5]), fmaxf(acc[6], acc[7]));
  pa = fmaxf(pa + cb, 0.f);               // bias+relu commute with max
  pb = fmaxf(pb + cb, 0.f);

  int pA = tileP * 4 + (half ? 2 : 0);
  int pB = pA + 1;
  h16* dst = h2 + (size_t)b * D1 + (size_t)n * 169;
  if (pA < 169) dst[pA] = (h16)pa;
  if (pB < 169) dst[pB] = (h16)pb;
}

// ---------------------------------------------------------------------------
// K4: gating layer 1: logits = h2 @ w_gate1, top-2 softmax gates,
// importance/load atomics, gates written to d_out (gt_l1) + ws.
// One wave per batch row.
// ---------------------------------------------------------------------------
__global__ void k_gate1(const h16* __restrict__ h2, const float* __restrict__ wg,
                        float* __restrict__ gates, float* __restrict__ gtOut,
                        float* __restrict__ accs) {
  int wave = threadIdx.x >> 5, lane = threadIdx.x & 31;
  int b = blockIdx.x * 4 + wave;
  const h16* xr = h2 + (size_t)b * D1;
  float acc[8] = {0, 0, 0, 0, 0, 0, 0, 0};
  for (int kk = lane; kk < D1; kk += 32) {
    float xv = (float)xr[kk];
    const float* wr = wg + (size_t)kk * 8;
#pragma unroll
    for (int e = 0; e < 8; ++e) acc[e] += xv * wr[e];
  }
#pragma unroll
  for (int e = 0; e < 8; ++e) acc[e] = waveAllSum(acc[e]);

  int i1 = 0; float v1 = acc[0];
#pragma unroll
  for (int e = 1; e < 8; ++e) if (acc[e] > v1) { v1 = acc[e]; i1 = e; }
  int i2 = (i1 == 0) ? 1 : 0; float v2 = acc[i2];
#pragma unroll
  for (int e = 0; e < 8; ++e)
    if (e != i1 && acc[e] > v2) { v2 = acc[e]; i2 = e; }

  float e2 = expf(v2 - v1);
  float inv = 1.f / (1.f + e2);
  float g1 = inv, g2 = e2 * inv;

  if (lane == 0) {
    float gv[8] = {0, 0, 0, 0, 0, 0, 0, 0};
    gv[i1] = g1; gv[i2] = g2;
#pragma unroll
    for (int e = 0; e < 8; ++e) {
      gates[b * 8 + e] = gv[e];
      gtOut[b * 8 + e] = gv[e];
    }
    atomicAdd(&accs[i1], g1);
    atomicAdd(&accs[i2], g2);
    atomicAdd(&accs[8 + i1], 1.f);
    atomicAdd(&accs[8 + i2], 1.f);
  }
}

// ---------------------------------------------------------------------------
// K5: dense all-expert GEMM  Eout[4096][1024] = h2[4096][10816] x W1t^T
// 4 waves stacked in M share one 64-col B panel: the 64x32 B tile is staged in
// LDS with double-buffered GLOBAL_LOAD_ASYNC_TO_LDS_B128 (ASYNCcnt), issued
// one K-step ahead; fragments then come from ds_load_b128.  A stays in global
// (rows are wave-private, served by L0).
// ---------------------------------------------------------------------------
__global__ void k_gemm1(const h16* __restrict__ A, const h16* __restrict__ B,
                        float* __restrict__ C) {
  __shared__ h16 sB[2][64 * 32];          // 2 x 4KB double buffer

  int tid  = threadIdx.x;
  int wave = tid >> 5, lane = tid & 31;
  int half = lane >> 4, lr = lane & 15;
  int m0 = blockIdx.y * 64 + wave * 16;
  int n0 = blockIdx.x * 64;

  const h16* ap = A + (size_t)(m0 + lr) * D1 + half * 8;

  // Each thread copies two 16B chunks of the 4KB tile: chunk = (row, quarter)
  int ch0 = tid,       r0 = ch0 >> 2, q0 = ch0 & 3;
  int ch1 = tid + 128, r1 = ch1 >> 2, q1 = ch1 & 3;
  const h16* g0 = B + (size_t)(n0 + r0) * D1 + q0 * 8;
  const h16* g1 = B + (size_t)(n0 + r1) * D1 + q1 * 8;
  unsigned l0[2], l1[2];
#pragma unroll
  for (int bi = 0; bi < 2; ++bi) {
    l0[bi] = (unsigned)(size_t)&sB[bi][r0 * 32 + q0 * 8];
    l1[bi] = (unsigned)(size_t)&sB[bi][r1 * 32 + q1 * 8];
  }

  v8f acc0 = {}, acc1 = {}, acc2 = {}, acc3 = {};

  // prologue: stage K-step 0 into buffer 0
  async_copy_b128(l0[0], g0);
  async_copy_b128(l1[0], g1);

  const int NK = D1 / 32;                 // 338
  for (int i = 0; i < NK; ++i) {
    int cur = i & 1;
    wait_asynccnt0();                     // own chunks of buf[cur] landed
    __syncthreads();                      // tile complete; buf[cur^1] free
    if (i + 1 < NK) {                     // stage next K-step into other buffer
      int nxt = cur ^ 1;
      const h16* gk = g0 + (size_t)(i + 1) * 32;
      async_copy_b128(l0[nxt], gk);
      async_copy_b128(l1[nxt], g1 + (size_t)(i + 1) * 32);
      (void)gk;
    }
    v16h a = ldfrag(ap + (size_t)i * 32);
    const h16* bb = &sB[cur][lr * 32 + half * 8];
    acc0 = wmma_f32_16x16x32(a, ldfrag(bb),           acc0);
    acc1 = wmma_f32_16x16x32(a, ldfrag(bb + 16 * 32), acc1);
    acc2 = wmma_f32_16x16x32(a, ldfrag(bb + 32 * 32), acc2);
    acc3 = wmma_f32_16x16x32(a, ldfrag(bb + 48 * 32), acc3);
  }

  int mbase = m0 + half * 8;
#pragma unroll
  for (int r = 0; r < 8; ++r) {
    float* crow = C + (size_t)(mbase + r) * 1024 + n0 + lr;
    crow[0]  = acc0[r];
    crow[16] = acc1[r];
    crow[32] = acc2[r];
    crow[48] = acc3[r];
  }
}

// ---------------------------------------------------------------------------
// K6: combine experts with gates + bias + relu -> y1[4096][128] f32
// ---------------------------------------------------------------------------
__global__ void k_combine1(const float* __restrict__ Eout,
                           const float* __restrict__ gates,
                           const float* __restrict__ b1, float* __restrict__ y1) {
  int idx = blockIdx.x * blockDim.x + threadIdx.x;  // 4096*128
  int o = idx & 127, b = idx >> 7;
  const float* er = Eout + (size_t)b * 1024;
  const float* gr = gates + b * 8;
  float y = 0.f;
#pragma unroll
  for (int e = 0; e < 8; ++e) {
    float g = gr[e];
    if (g != 0.f) y += g * (er[e * 128 + o] + b1[e * 128 + o]);
  }
  y1[idx] = fmaxf(y, 0.f);
}

// ---------------------------------------------------------------------------
// K7: MoE layer 2 (gating + top-2 experts + bias) + log_softmax.
// One wave per row; all reductions broadcast so every lane holds the result.
// ---------------------------------------------------------------------------
__global__ void k_moe2(const float* __restrict__ y1, const float* __restrict__ wg2,
                       const float* __restrict__ W2, const float* __restrict__ b2,
                       float* __restrict__ gtOut, float* __restrict__ accs,
                       float* __restrict__ logp) {
  int wave = threadIdx.x >> 5, lane = threadIdx.x & 31;
  int b = blockIdx.x * 4 + wave;

  float y[4];
#pragma unroll
  for (int j = 0; j < 4; ++j) y[j] = y1[(size_t)b * 128 + lane + 32 * j];

  float lg[8];
#pragma unroll
  for (int e = 0; e < 8; ++e) {
    float p = 0.f;
#pragma unroll
    for (int j = 0; j < 4; ++j) p += y[j] * wg2[(lane + 32 * j) * 8 + e];
    lg[e] = waveAllSum(p);
  }

  int i1 = 0; float v1 = lg[0];
#pragma unroll
  for (int e = 1; e < 8; ++e) if (lg[e] > v1) { v1 = lg[e]; i1 = e; }
  int i2 = (i1 == 0) ? 1 : 0; float v2 = lg[i2];
#pragma unroll
  for (int e = 0; e < 8; ++e)
    if (e != i1 && lg[e] > v2) { v2 = lg[e]; i2 = e; }

  float e2x = expf(v2 - v1);
  float inv = 1.f / (1.f + e2x);
  float g1 = inv, g2 = e2x * inv;

  if (lane == 0) {
    float gv[8] = {0, 0, 0, 0, 0, 0, 0, 0};
    gv[i1] = g1; gv[i2] = g2;
#pragma unroll
    for (int e = 0; e < 8; ++e) gtOut[b * 8 + e] = gv[e];
    atomicAdd(&accs[16 + i1], g1);
    atomicAdd(&accs[16 + i2], g2);
    atomicAdd(&accs[24 + i1], 1.f);
    atomicAdd(&accs[24 + i2], 1.f);
  }

  float ho[10];
#pragma unroll
  for (int o = 0; o < NOUT; ++o) {
    float p1 = 0.f, p2 = 0.f;
#pragma unroll
    for (int j = 0; j < 4; ++j) {
      int i = lane + 32 * j;
      p1 += y[j] * W2[((size_t)i1 * 128 + i) * NOUT + o];
      p2 += y[j] * W2[((size_t)i2 * 128 + i) * NOUT + o];
    }
    float s1 = waveAllSum(p1);
    float s2 = waveAllSum(p2);
    ho[o] = g1 * (s1 + b2[i1 * NOUT + o]) + g2 * (s2 + b2[i2 * NOUT + o]);
  }

  float mx = ho[0];
#pragma unroll
  for (int o = 1; o < NOUT; ++o) mx = fmaxf(mx, ho[o]);
  float s = 0.f;
#pragma unroll
  for (int o = 0; o < NOUT; ++o) s += expf(ho[o] - mx);
  float lz = mx + logf(s);
  if (lane < NOUT) logp[(size_t)b * NOUT + lane] = ho[lane] - lz;
}

// ---------------------------------------------------------------------------
// K8: cv^2 loss from the 4 accumulator groups
// ---------------------------------------------------------------------------
__global__ void k_loss(const float* __restrict__ accs, float* __restrict__ out) {
  if (threadIdx.x != 0 || blockIdx.x != 0) return;
  float total = 0.f;
#pragma unroll
  for (int g = 0; g < 4; ++g) {
    const float* v = accs + g * 8;
    float m = 0.f;
    for (int e = 0; e < 8; ++e) m += v[e];
    m *= 0.125f;
    float var = 0.f;
    for (int e = 0; e < 8; ++e) { float d = v[e] - m; var += d * d; }
    var *= (1.f / 7.f);                  // ddof=1
    total += var / (m * m + 1e-10f);
  }
  out[0] = total * 3e-5f;
}

// ---------------------------------------------------------------------------
extern "C" void kernel_launch(void* const* d_in, const int* in_sizes, int n_in,
                              void* d_out, int out_size, void* d_ws, size_t ws_size,
                              hipStream_t stream) {
  (void)in_sizes; (void)n_in; (void)out_size; (void)ws_size;
  const float* x   = (const float*)d_in[0];
  const float* c1w = (const float*)d_in[1];
  const float* c1b = (const float*)d_in[2];
  const float* c2w = (const float*)d_in[3];
  const float* c2b = (const float*)d_in[4];
  const float* wg1 = (const float*)d_in[5];
  const float* W1  = (const float*)d_in[6];
  const float* b1  = (const float*)d_in[7];
  const float* wg2 = (const float*)d_in[8];
  const float* W2  = (const float*)d_in[9];
  const float* b2  = (const float*)d_in[10];

  char* ws = (char*)d_ws;
  float* accs = (float*)(ws + WS_ACC);
  h16*   bmat = (h16*)(ws + WS_BMAT);
  h16*   w1t  = (h16*)(ws + WS_W1T);
  h16*   h1   = (h16*)(ws + WS_H1);
  h16*   h2   = (h16*)(ws + WS_H2);
  float* eout = (float*)(ws + WS_EOUT);
  float* gts1 = (float*)(ws + WS_GTS1);
  float* y1   = (float*)(ws + WS_Y1);

  float* out   = (float*)d_out;
  float* logp  = out;             // [4096][10]
  float* gt1o  = out + 40960;     // [4096][8]
  float* gt2o  = out + 73728;     // [4096][8]
  float* lossO = out + 106496;    // scalar

  k_zero_acc<<<1, 32, 0, stream>>>(accs);
  k_conv1<<<(BATCH * 784 * 32) / 256, 256, 0, stream>>>(x, c1w, c1b, h1);
  k_prep_bmat<<<(64 * 288 + 255) / 256, 256, 0, stream>>>(c2w, bmat);
  k_prep_w1t<<<(1024 * D1) / 256, 256, 0, stream>>>(W1, w1t);
  k_conv2pool<<<dim3(43, BATCH), 128, 0, stream>>>(h1, bmat, c2b, h2);
  k_gate1<<<BATCH / 4, 128, 0, stream>>>(h2, wg1, gts1, gt1o, accs);
  k_gemm1<<<dim3(16, BATCH / 64), 128, 0, stream>>>(h2, w1t, eout);
  k_combine1<<<(BATCH * 128) / 256, 256, 0, stream>>>(eout, gts1, b1, y1);
  k_moe2<<<BATCH / 4, 128, 0, stream>>>(y1, wg2, W2, b2, gt2o, accs, logp);
  k_loss<<<1, 32, 0, stream>>>(accs, lossO);
}
